// PredictionModel_53730040873516
// MI455X (gfx1250) — matrix operand order repaired and checked by reading, not compiled, not run on hardware
//
#include <hip/hip_runtime.h>
#include <hip/hip_bf16.h>
#include <stdint.h>

typedef __attribute__((ext_vector_type(16))) _Float16 v16h;
typedef __attribute__((ext_vector_type(8)))  _Float16 v8h;
typedef __attribute__((ext_vector_type(8)))  float    v8f;

#define WMMA_F16(a, b, c) \
  __builtin_amdgcn_wmma_f32_16x16x32_f16(false, (a), false, (b), (short)0, (c), false, false)

#define CAT16(lo, hi) \
  __builtin_shufflevector((lo), (hi), 0, 1, 2, 3, 4, 5, 6, 7, 8, 9, 10, 11, 12, 13, 14, 15)

// ---------------------------------------------------------------------------
// Weight repack: f32 (K x N) row-major -> WMMA B-operand packed f16.
//   i = ((kt*NT + nt)*32 + lane)*16 + h  ->  k = kt*32+(lane>>4)*8+(h&7)+(h>>3)*16,
//                                            n = nt*16+(lane&15)
// ---------------------------------------------------------------------------
__global__ void pack_w(const float* __restrict__ W, _Float16* __restrict__ out,
                       int K, int N, int srcN, int colOff, int Kp, int Np) {
  int i = blockIdx.x * blockDim.x + threadIdx.x;
  int total = Kp * Np;
  if (i >= total) return;
  int h    = i & 15;
  int lane = (i >> 4) & 31;
  int tile = i >> 9;
  int NT = Np >> 4;
  int nt = tile % NT;
  int kt = tile / NT;
  int k = kt * 32 + (lane >> 4) * 8 + (h & 7) + (h >> 3) * 16;
  int n = nt * 16 + (lane & 15);
  float v = (k < K && n < N) ? W[(size_t)k * srcN + colOff + n] : 0.0f;
  out[i] = (_Float16)v;
}

// ---------------------------------------------------------------------------
// Pack K rows of qkv into B-fragment layout per (head, 16-key tile, d-step).
//   i = (((head*(L/16)+kt)*2+ds)*32+lane)*16+h
// ---------------------------------------------------------------------------
__global__ void pack_k(const _Float16* __restrict__ qkv, _Float16* __restrict__ Kpk,
                       int L) {
  int i = blockIdx.x * blockDim.x + threadIdx.x;
  int total = 8 * (L >> 4) * 2 * 512;
  if (i >= total) return;
  int h = i & 15;
  int lane = (i >> 4) & 31;
  int rest = i >> 9;
  int ds = rest & 1; rest >>= 1;
  int kt = rest % (L >> 4);
  int head = rest / (L >> 4);
  int key = kt * 16 + (lane & 15);
  int d = ds * 32 + (lane >> 4) * 8 + (h & 7) + (h >> 3) * 16;
  Kpk[i] = qkv[(size_t)key * 1536 + 512 + head * 64 + d];
}

// ---------------------------------------------------------------------------
// Pack V rows of qkv into B-fragment layout per (head, 32-key block, d-tile).
//   i = (((head*(L/32)+vb)*4+t)*32+lane)*16+h
// ---------------------------------------------------------------------------
__global__ void pack_v(const _Float16* __restrict__ qkv, _Float16* __restrict__ Vpk,
                       int L) {
  int i = blockIdx.x * blockDim.x + threadIdx.x;
  int total = 8 * (L >> 5) * 4 * 512;
  if (i >= total) return;
  int h = i & 15;
  int lane = (i >> 4) & 31;
  int rest = i >> 9;
  int t = rest & 3; rest >>= 2;
  int vb = rest % (L >> 5);
  int head = rest / (L >> 5);
  int key = vb * 32 + (lane >> 4) * 8 + (h & 7) + (h >> 3) * 16;
  int n = t * 16 + (lane & 15);
  Vpk[i] = qkv[(size_t)key * 1536 + 1024 + head * 64 + n];
}

// ---------------------------------------------------------------------------
// Per-point MLP: pts (B,16,4) -> feats (B, 256) f16.  (~80 MFLOP total)
// ---------------------------------------------------------------------------
__global__ void point_mlp(const float* __restrict__ x,
                          const float* __restrict__ W1, const float* __restrict__ b1,
                          const float* __restrict__ g1, const float* __restrict__ be1,
                          const float* __restrict__ W2, const float* __restrict__ b2,
                          const float* __restrict__ g2, const float* __restrict__ be2,
                          _Float16* __restrict__ feats, int B) {
  int b = blockIdx.x * blockDim.x + threadIdx.x;
  if (b >= B) return;
  const float* pt = x + (size_t)b * 88 + 22;
  for (int p = 0; p < 16; ++p) {
    float h1[32];
    const float* w1 = W1 + p * 4 * 32;
    float m = 0.f;
    for (int j = 0; j < 32; ++j) {
      float s = b1[p * 32 + j];
      for (int d = 0; d < 4; ++d) s += pt[p * 4 + d] * w1[d * 32 + j];
      h1[j] = s; m += s;
    }
    m *= (1.f / 32.f);
    float v = 0.f;
    for (int j = 0; j < 32; ++j) { float d = h1[j] - m; v += d * d; }
    float rinv = rsqrtf(v * (1.f / 32.f) + 1e-5f);
    for (int j = 0; j < 32; ++j) {
      float y = (h1[j] - m) * rinv * g1[p * 32 + j] + be1[p * 32 + j];
      h1[j] = y >= 0.f ? y : 0.2f * y;
    }
    float h2[16];
    const float* w2 = W2 + p * 32 * 16;
    float m2 = 0.f;
    for (int k = 0; k < 16; ++k) {
      float s = b2[p * 16 + k];
      for (int j = 0; j < 32; ++j) s += h1[j] * w2[j * 16 + k];
      h2[k] = s; m2 += s;
    }
    m2 *= (1.f / 16.f);
    float v2 = 0.f;
    for (int k = 0; k < 16; ++k) { float d = h2[k] - m2; v2 += d * d; }
    float rinv2 = rsqrtf(v2 * (1.f / 16.f) + 1e-5f);
    for (int k = 0; k < 16; ++k) {
      float y = (h2[k] - m2) * rinv2 * g2[p * 16 + k] + be2[p * 16 + k];
      y = y >= 0.f ? y : 0.2f * y;
      feats[(size_t)b * 256 + p * 16 + k] = (_Float16)y;
    }
  }
}

// ---------------------------------------------------------------------------
// Fused WMMA GEMM: Y = epilogue(X @ W + bias).
//   modes: 0 = bias; 1 = bias+LN; 2 = bias+LN+lrelu; 3 = bias+LN+res+lrelu
// Block = 16 rows x Np cols; 4 waves; TPW column tiles/wave (Np = TPW*64).
// ---------------------------------------------------------------------------
template <int TPW>
__global__ __launch_bounds__(128) void gemm_fused(
    const _Float16* __restrict__ X, int Kp,
    const _Float16* __restrict__ Wp,
    const float* __restrict__ bias,
    const float* __restrict__ gamma,
    const float* __restrict__ beta,
    const _Float16* __restrict__ residual,
    _Float16* __restrict__ Y, int ldY, int colOff, int mode) {
  constexpr int NT = TPW * 4;
  constexpr int Np = NT * 16;
  __shared__ __align__(16) _Float16 sX[16 * 512];
  __shared__ float sOut[16 * Np];
  __shared__ float sRed[16 * 8];
  __shared__ float sRed2[16 * 8];
  __shared__ float sMean[16];
  __shared__ float sRinv[16];

  const int tid = threadIdx.x;
  const int lane = tid & 31;
  const int wave = tid >> 5;
  const int row0 = blockIdx.x * 16;

  {  // stage 16 x Kp A-tile in LDS (rows are contiguous in X): b128 copies
    const uint4* src = (const uint4*)(X + (size_t)row0 * Kp);
    uint4* dst = (uint4*)sX;
    const int total = 2 * Kp;
    for (int i = tid; i < total; i += 128) dst[i] = src[i];
  }
  __syncthreads();

  const int m = lane & 15;
  const int kb = (lane >> 4) * 8;
  const int ksteps = Kp >> 5;

  auto ldA = [&](int ks) -> v16h {
    const _Float16* ax = sX + m * Kp + ks * 32 + kb;
    v8h lo = *(const v8h*)ax;
    v8h hh = *(const v8h*)(ax + 16);
    return CAT16(lo, hh);
  };

  v8f acc[TPW] = {};
  v16h a = ldA(0);
  for (int ks = 0; ks < ksteps; ++ks) {
    const _Float16* wb = Wp + (((size_t)ks * NT + wave * TPW) * 32 + lane) * 16;
    v16h bfr[TPW];
#pragma unroll
    for (int t = 0; t < TPW; ++t) bfr[t] = *(const v16h*)(wb + (size_t)t * 512);
    v16h an;
    if (ks + 1 < ksteps) an = ldA(ks + 1);
#pragma unroll
    for (int t = 0; t < TPW; ++t) acc[t] = WMMA_F16(a, bfr[t], acc[t]);
    if (ks + 1 < ksteps) a = an;
  }

  const int ro = (lane >> 4) * 8;
#pragma unroll
  for (int t = 0; t < TPW; ++t) {
    const int nc = (wave * TPW + t) * 16 + (lane & 15);
    const float bv = bias ? bias[nc] : 0.0f;
#pragma unroll
    for (int r = 0; r < 8; ++r) sOut[(ro + r) * Np + nc] = acc[t][r] + bv;
  }
  __syncthreads();

  if (mode >= 1) {
    const int row = tid & 15;
    const int slot = tid >> 4;  // 0..7
    float s = 0.f, ss = 0.f;
    for (int c = slot; c < Np; c += 8) { float v = sOut[row * Np + c]; s += v; ss += v * v; }
    sRed[row * 8 + slot] = s;
    sRed2[row * 8 + slot] = ss;
    __syncthreads();
    if (slot == 0) {
      float s1 = 0.f, s2 = 0.f;
      for (int p = 0; p < 8; ++p) { s1 += sRed[row * 8 + p]; s2 += sRed2[row * 8 + p]; }
      const float mean = s1 / Np;
      const float var = s2 / Np - mean * mean;
      sMean[row] = mean;
      sRinv[row] = rsqrtf(var + 1e-5f);
    }
    __syncthreads();
  }
  // coalesced epilogue: consecutive lanes -> consecutive columns
  for (int idx = tid; idx < 16 * Np; idx += 128) {
    const int row = idx / Np;
    const int c = idx - row * Np;
    float v = sOut[idx];
    if (mode >= 1) v = (v - sMean[row]) * sRinv[row] * gamma[c] + beta[c];
    if (mode == 3) v += (float)residual[(size_t)(row0 + row) * ldY + colOff + c];
    if (mode >= 2) v = v >= 0.f ? v : 0.2f * v;
    Y[(size_t)(row0 + row) * ldY + colOff + c] = (_Float16)v;
  }
}

// ---------------------------------------------------------------------------
// Flash attention, 8 heads, d=64, online softmax, packed K/V fragments.
// 1 wave per (head, 16-query tile).  K/V streams are linear & prefetched.
// ---------------------------------------------------------------------------
__global__ __launch_bounds__(32) void flash_attn(const _Float16* __restrict__ qkv,
                                                 const _Float16* __restrict__ Kpk,
                                                 const _Float16* __restrict__ Vpk,
                                                 _Float16* __restrict__ obuf, int L) {
  const int head = blockIdx.y;
  const int q0 = blockIdx.x * 16;
  const int lane = threadIdx.x;
  const int m = lane & 15;
  const int hi = lane >> 4;
  const int ro = hi * 8;
  const int nvb = L >> 5;

  __shared__ __align__(16) float sS[16 * 32];
  __shared__ __align__(16) _Float16 sP[16 * 32];
  __shared__ float sRed[32];
  __shared__ float sCorr[16];

  // Q fragments (scaled by 1/sqrt(64))
  v16h qa0, qa1;
  {
    const _Float16* qrow = qkv + (size_t)(q0 + m) * 1536 + head * 64 + hi * 8;
    v8h l0 = *(const v8h*)(qrow);
    v8h h0 = *(const v8h*)(qrow + 16);
    v8h l1 = *(const v8h*)(qrow + 32);
    v8h h1 = *(const v8h*)(qrow + 48);
    qa0 = CAT16(l0, h0);
    qa1 = CAT16(l1, h1);
#pragma unroll
    for (int h = 0; h < 16; ++h) {
      qa0[h] = qa0[h] * (_Float16)0.125f;
      qa1[h] = qa1[h] * (_Float16)0.125f;
    }
  }

  const _Float16* kfrag = Kpk + ((size_t)head * (L >> 4) * 2) * 512 + lane * 16;
  const _Float16* vfrag = Vpk + ((size_t)head * (L >> 5) * 4) * 512 + lane * 16;

  v16h k00 = *(const v16h*)(kfrag);
  v16h k01 = *(const v16h*)(kfrag + 512);
  v16h k10 = *(const v16h*)(kfrag + 1024);
  v16h k11 = *(const v16h*)(kfrag + 1536);

  v8f o0 = {}, o1 = {}, o2 = {}, o3 = {};
  float mrun = -3e38f, lrun = 0.f;

  for (int vb = 0; vb < nvb; ++vb) {
    v8f s0 = {}, s1 = {};
    s0 = WMMA_F16(qa0, k00, s0);
    s0 = WMMA_F16(qa1, k01, s0);
    s1 = WMMA_F16(qa0, k10, s1);
    s1 = WMMA_F16(qa1, k11, s1);

    // preload next key-block K fragments; load current V fragments
    if (vb + 1 < nvb) {
      const _Float16* nk = kfrag + (size_t)(vb + 1) * 2048;
      k00 = *(const v16h*)(nk);
      k01 = *(const v16h*)(nk + 512);
      k10 = *(const v16h*)(nk + 1024);
      k11 = *(const v16h*)(nk + 1536);
    }
    const _Float16* vk = vfrag + (size_t)vb * 2048;
    v16h bv0 = *(const v16h*)(vk);
    v16h bv1 = *(const v16h*)(vk + 512);
    v16h bv2 = *(const v16h*)(vk + 1024);
    v16h bv3 = *(const v16h*)(vk + 1536);
    if (vb + 2 < nvb) {  // gfx1250 global_prefetch two blocks ahead
      __builtin_prefetch((const void*)(kfrag + (size_t)(vb + 2) * 2048), 0, 1);
      __builtin_prefetch((const void*)(vfrag + (size_t)(vb + 2) * 2048), 0, 1);
    }

#pragma unroll
    for (int r = 0; r < 8; ++r) {
      sS[(ro + r) * 32 + m] = s0[r];
      sS[(ro + r) * 32 + 16 + m] = s1[r];
    }
    __syncthreads();

    // parallel online softmax: lane owns (row m, column half hi)
    const float* srow = sS + m * 32 + hi * 16;
    float pmax = srow[0];
#pragma unroll
    for (int j = 1; j < 16; ++j) pmax = fmaxf(pmax, srow[j]);
    sRed[lane] = pmax;
    __syncthreads();
    const float mx = fmaxf(mrun, fmaxf(sRed[m], sRed[m + 16]));
    const float corr = __expf(mrun - mx);
    _Float16* prow = sP + m * 32 + hi * 16;
    float psum = 0.f;
#pragma unroll
    for (int j = 0; j < 16; ++j) {
      const float p = __expf(srow[j] - mx);
      prow[j] = (_Float16)p;
      psum += p;
    }
    sRed[lane] = psum;
    if (hi == 0) sCorr[m] = corr;
    __syncthreads();
    lrun = lrun * corr + sRed[m] + sRed[m + 16];
    mrun = mx;

    // rescale O accumulators by per-row correction
#pragma unroll
    for (int r = 0; r < 8; ++r) {
      const float c = sCorr[ro + r];
      o0[r] *= c; o1[r] *= c; o2[r] *= c; o3[r] *= c;
    }

    // P fragment (A operand) from LDS: two b128 chunks
    v16h pa;
    {
      const _Float16* px = sP + m * 32 + hi * 8;
      v8h lo = *(const v8h*)px;
      v8h hh = *(const v8h*)(px + 16);
      pa = CAT16(lo, hh);
    }
    o0 = WMMA_F16(pa, bv0, o0);
    o1 = WMMA_F16(pa, bv1, o1);
    o2 = WMMA_F16(pa, bv2, o2);
    o3 = WMMA_F16(pa, bv3, o3);
    __syncthreads();
  }

  if (hi == 0) sCorr[m] = 1.0f / lrun;
  __syncthreads();
#pragma unroll
  for (int r = 0; r < 8; ++r) {
    const float c = sCorr[ro + r];
    const size_t orow = (size_t)(q0 + ro + r) * 512 + head * 64 + m;
    obuf[orow]      = (_Float16)(o0[r] * c);
    obuf[orow + 16] = (_Float16)(o1[r] * c);
    obuf[orow + 32] = (_Float16)(o2[r] * c);
    obuf[orow + 48] = (_Float16)(o3[r] * c);
  }
}

// ---------------------------------------------------------------------------
// Build fu input: concat(measurements[22], f[128]) zero-padded to 160, f16.
// ---------------------------------------------------------------------------
__global__ void build_fu_in(const float* __restrict__ x, const _Float16* __restrict__ f,
                            _Float16* __restrict__ out, int B) {
  int i = blockIdx.x * blockDim.x + threadIdx.x;
  if (i >= B * 160) return;
  const int b = i / 160, c = i % 160;
  _Float16 v;
  if (c < 22)       v = (_Float16)x[(size_t)b * 88 + c];
  else if (c < 150) v = f[(size_t)b * 128 + (c - 22)];
  else              v = (_Float16)0.f;
  out[i] = v;
}

// ---------------------------------------------------------------------------
// Final head: out = tanh(F @ fin_W + fin_b) * scale + target + biasf  (f32)
// ---------------------------------------------------------------------------
__global__ __launch_bounds__(32) void final_head(const _Float16* __restrict__ F,
                                                 const _Float16* __restrict__ Wp,
                                                 const float* __restrict__ fb,
                                                 const float* __restrict__ scale,
                                                 const float* __restrict__ biasf,
                                                 const float* __restrict__ x,
                                                 float* __restrict__ out) {
  __shared__ __align__(16) _Float16 sX[16 * 128];
  __shared__ float sOut[16 * 32];
  const int lane = threadIdx.x;
  const int row0 = blockIdx.x * 16;
  {
    const uint4* src = (const uint4*)(F + (size_t)row0 * 128);
    uint4* dst = (uint4*)sX;
    for (int i = lane; i < 256; i += 32) dst[i] = src[i];
  }
  __syncthreads();
  const int m = lane & 15, hi = lane >> 4;
  v8f a0 = {}, a1 = {};
  for (int ks = 0; ks < 4; ++ks) {
    const _Float16* ax = sX + m * 128 + ks * 32 + hi * 8;
    v8h lo = *(const v8h*)ax;
    v8h hh = *(const v8h*)(ax + 16);
    v16h a = CAT16(lo, hh);
    const v16h b0 = *(const v16h*)(Wp + (((size_t)ks * 2 + 0) * 32 + lane) * 16);
    const v16h b1 = *(const v16h*)(Wp + (((size_t)ks * 2 + 1) * 32 + lane) * 16);
    a0 = WMMA_F16(a, b0, a0);
    a1 = WMMA_F16(a, b1, a1);
  }
  const int ro = hi * 8;
#pragma unroll
  for (int r = 0; r < 8; ++r) {
    sOut[(ro + r) * 32 + m] = a0[r];
    sOut[(ro + r) * 32 + 16 + m] = a1[r];
  }
  __syncthreads();
  for (int i = lane; i < 16 * 21; i += 32) {
    const int row = i / 21, c = i % 21;
    const float v = tanhf(sOut[row * 32 + c] + fb[c]);
    out[(size_t)(row0 + row) * 21 + c] =
        v * scale[c] + x[(size_t)(row0 + row) * 88 + c] + biasf[c];
  }
}

// ---------------------------------------------------------------------------
// Host launcher
// ---------------------------------------------------------------------------
extern "C" void kernel_launch(void* const* d_in, const int* in_sizes, int n_in,
                              void* d_out, int out_size, void* d_ws, size_t ws_size,
                              hipStream_t stream) {
  const float* x      = (const float*)d_in[0];
  const float* pm_W1  = (const float*)d_in[1];
  const float* pm_b1  = (const float*)d_in[2];
  const float* pm_g1  = (const float*)d_in[3];
  const float* pm_be1 = (const float*)d_in[4];
  const float* pm_W2  = (const float*)d_in[5];
  const float* pm_b2  = (const float*)d_in[6];
  const float* pm_g2  = (const float*)d_in[7];
  const float* pm_be2 = (const float*)d_in[8];
  const float* ff_W   = (const float*)d_in[9];
  const float* ff_b   = (const float*)d_in[10];
  const float* ff_g   = (const float*)d_in[11];
  const float* ff_be  = (const float*)d_in[12];
  const float* la_inW = (const float*)d_in[13];
  const float* la_inb = (const float*)d_in[14];
  const float* la_outW= (const float*)d_in[15];
  const float* la_outb= (const float*)d_in[16];
  const float* la_ng  = (const float*)d_in[17];
  const float* la_nb  = (const float*)d_in[18];
  const float* llf_W  = (const float*)d_in[19];
  const float* llf_b  = (const float*)d_in[20];
  const float* llf_g  = (const float*)d_in[21];
  const float* llf_be = (const float*)d_in[22];
  const float* fl_W   = (const float*)d_in[23];
  const float* fl_b   = (const float*)d_in[24];
  const float* fl_g   = (const float*)d_in[25];
  const float* fl_be  = (const float*)d_in[26];
  const float* fu_W   = (const float*)d_in[27];
  const float* fu_b   = (const float*)d_in[28];
  const float* fu_g   = (const float*)d_in[29];
  const float* fu_be  = (const float*)d_in[30];
  const float* ga_inW = (const float*)d_in[31];
  const float* ga_inb = (const float*)d_in[32];
  const float* ga_outW= (const float*)d_in[33];
  const float* ga_outb= (const float*)d_in[34];
  const float* ga_ng  = (const float*)d_in[35];
  const float* ga_nb  = (const float*)d_in[36];
  const float* rb_W1  = (const float*)d_in[37];
  const float* rb_b1  = (const float*)d_in[38];
  const float* rb_g1  = (const float*)d_in[39];
  const float* rb_be1 = (const float*)d_in[40];
  const float* rb_W2  = (const float*)d_in[41];
  const float* rb_b2  = (const float*)d_in[42];
  const float* rb_g2  = (const float*)d_in[43];
  const float* rb_be2 = (const float*)d_in[44];
  const float* fin_W  = (const float*)d_in[45];
  const float* fin_b  = (const float*)d_in[46];
  const float* scale  = (const float*)d_in[47];
  const float* biasf  = (const float*)d_in[48];
  float* out = (float*)d_out;

  const int B = in_sizes[0] / 88;  // 4096
  const int MB = B / 16;

  // ---- workspace layout (offsets in halves for packed region) ----
  _Float16* pk = (_Float16*)d_ws;
  size_t off = 0;
  const size_t PK_FF  = off; off += 256 * 512;
  const size_t PK_LAQ = off; off += 512 * 512;
  const size_t PK_LAK = off; off += 512 * 512;
  const size_t PK_LAV = off; off += 512 * 512;
  const size_t PK_LAO = off; off += 512 * 512;
  const size_t PK_LLF = off; off += 512 * 128;
  const size_t PK_FL  = off; off += 128 * 128;
  const size_t PK_FU  = off; off += 160 * 128;
  const size_t PK_GAV = off; off += 128 * 128;
  const size_t PK_GAO = off; off += 128 * 128;
  const size_t PK_RB1 = off; off += 3 * 128 * 128;
  const size_t PK_RB2 = off; off += 3 * 128 * 128;
  const size_t PK_FIN = off; off += 128 * 32;
  const size_t PK_K   = off; off += (size_t)8 * B * 64;  // packed K fragments
  const size_t PK_V   = off; off += (size_t)8 * B * 64;  // packed V fragments
  char* actBase = (char*)d_ws + ((off * 2 + 255) & ~(size_t)255);
  _Float16* qkvB = (_Float16*)(actBase);                                // B x 1536
  _Float16* buf1 = (_Float16*)(actBase + (size_t)B * 1536 * 2);         // B x 512
  _Float16* buf2 = (_Float16*)(actBase + (size_t)B * (1536 + 512) * 2); // B x 512
  _Float16* buf3 = (_Float16*)(actBase + (size_t)B * (1536 + 1024) * 2);// B x 160

  auto packw = [&](const float* W, size_t dstOff, int K, int N, int srcN,
                   int colOff, int Kp, int Np) {
    const int total = Kp * Np;
    pack_w<<<(total + 255) / 256, 256, 0, stream>>>(W, pk + dstOff, K, N, srcN,
                                                    colOff, Kp, Np);
  };
  packw(ff_W,    PK_FF,  256, 512, 512, 0,    256, 512);
  packw(la_inW,  PK_LAQ, 512, 512, 1536, 0,   512, 512);
  packw(la_inW,  PK_LAK, 512, 512, 1536, 512, 512, 512);
  packw(la_inW,  PK_LAV, 512, 512, 1536, 1024,512, 512);
  packw(la_outW, PK_LAO, 512, 512, 512, 0,    512, 512);
  packw(llf_W,   PK_LLF, 512, 128, 128, 0,    512, 128);
  packw(fl_W,    PK_FL,  128, 128, 128, 0,    128, 128);
  packw(fu_W,    PK_FU,  150, 128, 128, 0,    160, 128);
  packw(ga_inW,  PK_GAV, 128, 128, 384, 256,  128, 128);
  packw(ga_outW, PK_GAO, 128, 128, 128, 0,    128, 128);
  for (int i = 0; i < 3; ++i) {
    packw(rb_W1 + (size_t)i * 128 * 128, PK_RB1 + (size_t)i * 128 * 128,
          128, 128, 128, 0, 128, 128);
    packw(rb_W2 + (size_t)i * 128 * 128, PK_RB2 + (size_t)i * 128 * 128,
          128, 128, 128, 0, 128, 128);
  }
  packw(fin_W, PK_FIN, 128, 21, 21, 0, 128, 32);

  // 1) point MLP -> feats in buf2 (B x 256)
  point_mlp<<<(B + 127) / 128, 128, 0, stream>>>(
      x, pm_W1, pm_b1, pm_g1, pm_be1, pm_W2, pm_b2, pm_g2, pm_be2, buf2, B);

  // 2) pf = lrelu(LN(feats @ ff_W + b))  -> buf1 (B x 512)
  gemm_fused<8><<<MB, 128, 0, stream>>>(buf2, 256, pk + PK_FF, ff_b, ff_g, ff_be,
                                        nullptr, buf1, 512, 0, 2);
  // 3) qkv = pf @ la_inW + b  (three 512->512 slices into B x 1536)
  gemm_fused<8><<<MB, 128, 0, stream>>>(buf1, 512, pk + PK_LAQ, la_inb + 0,
                                        nullptr, nullptr, nullptr, qkvB, 1536, 0, 0);
  gemm_fused<8><<<MB, 128, 0, stream>>>(buf1, 512, pk + PK_LAK, la_inb + 512,
                                        nullptr, nullptr, nullptr, qkvB, 1536, 512, 0);
  gemm_fused<8><<<MB, 128, 0, stream>>>(buf1, 512, pk + PK_LAV, la_inb + 1024,
                                        nullptr, nullptr, nullptr, qkvB, 1536, 1024, 0);
  // 3b) pack K/V rows into WMMA B-fragment streams
  {
    const int totK = 8 * (B >> 4) * 2 * 512;
    pack_k<<<(totK + 255) / 256, 256, 0, stream>>>(qkvB, pk + PK_K, B);
    const int totV = 8 * (B >> 5) * 4 * 512;
    pack_v<<<(totV + 255) / 256, 256, 0, stream>>>(qkvB, pk + PK_V, B);
  }
  // 4) flash attention -> buf2 (B x 512)
  {
    dim3 grid(B / 16, 8);
    flash_attn<<<grid, 32, 0, stream>>>(qkvB, pk + PK_K, pk + PK_V, buf2, B);
  }
  // 5) llf = LN(o @ la_outW + b, la_ng, la_nb)  -> buf1
  gemm_fused<8><<<MB, 128, 0, stream>>>(buf2, 512, pk + PK_LAO, la_outb, la_ng,
                                        la_nb, nullptr, buf1, 512, 0, 1);
  // 6) llf = lrelu(LN(llf @ llf_W + b))  (512 -> 128) -> buf2
  gemm_fused<2><<<MB, 128, 0, stream>>>(buf1, 512, pk + PK_LLF, llf_b, llf_g,
                                        llf_be, nullptr, buf2, 128, 0, 2);
  // 7) f = lrelu(LN(llf @ fl_W + b))  -> buf1
  gemm_fused<2><<<MB, 128, 0, stream>>>(buf2, 128, pk + PK_FL, fl_b, fl_g,
                                        fl_be, nullptr, buf1, 128, 0, 2);
  // 8) comb = [measurements, f] -> buf3 (B x 160)
  build_fu_in<<<(B * 160 + 255) / 256, 256, 0, stream>>>(x, buf1, buf3, B);
  // 9) f = lrelu(LN(comb @ fu_W + b)) -> buf2
  gemm_fused<2><<<MB, 128, 0, stream>>>(buf3, 160, pk + PK_FU, fu_b, fu_g,
                                        fu_be, nullptr, buf2, 128, 0, 2);
  // 10) v = f @ ga_inW[:, 2H:] + b[2H:]  -> buf1
  gemm_fused<2><<<MB, 128, 0, stream>>>(buf2, 128, pk + PK_GAV, ga_inb + 256,
                                        nullptr, nullptr, nullptr, buf1, 128, 0, 0);
  // 11) f = LN(v @ ga_outW + b, ga_ng, ga_nb) -> buf2
  gemm_fused<2><<<MB, 128, 0, stream>>>(buf1, 128, pk + PK_GAO, ga_outb, ga_ng,
                                        ga_nb, nullptr, buf2, 128, 0, 1);
  // 12) three residual blocks (f lives in buf2)
  for (int i = 0; i < 3; ++i) {
    gemm_fused<2><<<MB, 128, 0, stream>>>(buf2, 128, pk + PK_RB1 + (size_t)i * 128 * 128,
                                          rb_b1 + i * 128, rb_g1 + i * 128,
                                          rb_be1 + i * 128, nullptr, buf1, 128, 0, 2);
    gemm_fused<2><<<MB, 128, 0, stream>>>(buf1, 128, pk + PK_RB2 + (size_t)i * 128 * 128,
                                          rb_b2 + i * 128, rb_g2 + i * 128,
                                          rb_be2 + i * 128, buf2, buf2, 128, 0, 3);
  }
  // 13) final head -> d_out (B x 21 f32)
  final_head<<<MB, 32, 0, stream>>>(buf2, pk + PK_FIN, fin_b, scale, biasf, x, out);
}
